// CP_sparse_degree_LU_32882269618787
// MI455X (gfx1250) — matrix operand order repaired and checked by reading, not compiled, not run on hardware
//
#include <hip/hip_runtime.h>
#include <hip/hip_bf16.h>

typedef __attribute__((ext_vector_type(16))) __bf16 v16bf;
typedef __attribute__((ext_vector_type(8)))  __bf16 v8bf;
typedef __attribute__((ext_vector_type(8)))  float  v8f;
typedef __attribute__((ext_vector_type(4)))  unsigned int v4u;
typedef __attribute__((ext_vector_type(8)))  int v8i;
typedef __attribute__((ext_vector_type(4)))  int v4i;

union BF16Frag { v16bf v; v8bf h[2]; };

static constexpr int  DDIM = 2048;     // D == K == O == 2048
static constexpr long NROW = 8192;     // 4 * 2048
static constexpr int  BM = 128, BN = 128, BK = 32;
static constexpr int  LDSP = 40;       // LDS row stride (80 B) = 64 B row + 16 B TDM pad
static constexpr int  TILE_ELEMS = BM * LDSP;          // 5120 bf16 per buffer
static constexpr unsigned TILE_BYTES = TILE_ELEMS * 2; // 10240 B
static constexpr int  NBUF = 3;        // 3-deep TDM pipeline (distance 2)

// ---------------- conversion pre-passes ----------------

__global__ void cvt_f32_to_bf16(const float* __restrict__ src,
                                __bf16* __restrict__ dst, long n) {
    long i = (long)blockIdx.x * blockDim.x + threadIdx.x;
    long stride = (long)gridDim.x * blockDim.x;
    for (; i < n; i += stride) dst[i] = (__bf16)src[i];
}

__global__ void cvt_mask_mul_bf16(const float* __restrict__ U,
                                  const float* __restrict__ masks,
                                  __bf16* __restrict__ dst, long n) {
    long i = (long)blockIdx.x * blockDim.x + threadIdx.x;
    long stride = (long)gridDim.x * blockDim.x;
    for (; i < n; i += stride) dst[i] = (__bf16)(U[i] * masks[i]);
}

// ---------------- TDM tile load: 128 rows x 32 bf16, row-padded to 80 B ----------------
// D# per CDNA5 ISA ch.8: group0 = {count, lds_addr, global_addr(57b), type=2}
// group1: data_size=2B, pad_enable, pad_interval=16 DW (pad after each 64 B row),
//         pad_amount=4 DW (16 B) -> LDS row stride 80 B; tensor==tile dims (no OOB),
//         tensor_dim0_stride = 2048 elements (source row pitch).
static __device__ __forceinline__
void tdm_load_tile(unsigned long long gaddr_bytes, unsigned lds_byte_off) {
    v4u g0 = {
        1u,                                           // count=1, user descriptor
        lds_byte_off,                                 // LDS byte address
        (unsigned)(gaddr_bytes & 0xffffffffu),        // global_addr[31:0]
        (unsigned)((gaddr_bytes >> 32) & 0x01ffffffu) // global_addr[56:32]
            | (2u << 30)                              // type = 2 ("image")
    };
    v8i g1 = {
        (int)((1u << 16)        // data_size = 1 -> 2 bytes
            | (1u << 20)        // pad_enable
            | (3u << 22)        // pad_interval: 16 DWORDs = 64 B
            | (3u << 25)),      // pad_amount:   4 DWORDs = 16 B
        (int)(32u  << 16),      // tensor_dim0 = 32   (bits 79:48)
        (int)(128u << 16),      // tensor_dim1 = 128  (bits 111:80)
        (int)(32u  << 16),      // tile_dim0 = 32     (bits 127:112)
        (int)(128u),            // tile_dim1 = 128, tile_dim2 = 0
        (int)(2048u),           // tensor_dim0_stride = 2048 elems (bits 207:160)
        0, 0                    // tensor_dim1_stride = 0 (2D)
    };
    v4i z4 = {0, 0, 0, 0};
    v8i z8 = {0, 0, 0, 0, 0, 0, 0, 0};
    // 6-arg variant (clang-23 / therock-10.0 headers)
    __builtin_amdgcn_tensor_load_to_lds(g0, g1, z4, z4, z8, 0);
}

// ---------------- fused WMMA GEMM ----------------
// C[m,n] = sum_k A[m,k] * B[n,k]   (both operands row-major, K contiguous)
// mode 0: Cf = acc
// mode 1: Cf = acc*Cf + Cf
// mode 2: Cf = acc*Cf + Cf; obf = bf16(Cf)
// mode 3: Xout = acc + bias[n]
__global__ __launch_bounds__(256)
void gemm_chain_bf16(const __bf16* __restrict__ A,
                     const __bf16* __restrict__ B,
                     float* __restrict__ Cf,
                     __bf16* __restrict__ obf,
                     const float* __restrict__ bias,
                     float* __restrict__ Xout,
                     int mode) {
    // [A buf0..2][B buf0..2], each 128x40 bf16 (10240 B) => 60 KB
    __shared__ __align__(128) __bf16 smem[2 * NBUF * TILE_ELEMS];

    const int tid   = threadIdx.x;
    const int lane  = tid & 31;
    const int wid   = tid >> 5;        // 8 waves
    const int wm    = wid & 1;         // 2 waves along M
    const int wn    = wid >> 1;        // 4 waves along N
    const int lrow  = lane & 15;
    const int khalf = lane >> 4;       // 0/1 per CDNA5 wave32 layout

    const int rowBase = blockIdx.y * BM;
    const int colBase = blockIdx.x * BN;

    // Escape smem into opaque asm with a memory clobber: the TDM engine writes
    // LDS behind the compiler's back, so the ds_loads below must not be folded.
    // Launder only an integer offset so pointer provenance stays addrspace(3)
    // and the reads lower to ds_load_b128 (not flat_load).
    unsigned lzero = 0;
    asm volatile("" : "+s"(lzero) : "s"(&smem[0]) : "memory");

    // Byte addresses of this block's tile rows in global memory.
    const unsigned long long aBase =
        (unsigned long long)(size_t)A + (unsigned long long)rowBase * DDIM * 2ull;
    const unsigned long long bBase =
        (unsigned long long)(size_t)B + (unsigned long long)colBase * DDIM * 2ull;

    const v8f vzero = {0.f, 0.f, 0.f, 0.f, 0.f, 0.f, 0.f, 0.f};
    v8f acc[4][2];
#pragma unroll
    for (int mt = 0; mt < 4; ++mt)
#pragma unroll
        for (int nt = 0; nt < 2; ++nt) acc[mt][nt] = vzero;

    constexpr int NK = DDIM / BK;      // 64 K-steps

    // Prologue: DMA tiles 0 and 1 into buffers 0 and 1 (A and B each).
    if (wid == 0) {
        tdm_load_tile(aBase, 0u);
        tdm_load_tile(bBase, NBUF * TILE_BYTES);
        tdm_load_tile(aBase + BK * 2ull, 1u * TILE_BYTES);
        tdm_load_tile(bBase + BK * 2ull, (NBUF + 1u) * TILE_BYTES);
    }

    int buf = 0;
    for (int kt = 0; kt < NK; ++kt) {
        if (wid == 0) {
            if (kt + 2 < NK) {
                // DMA tile kt+2 into the buffer retired at iteration kt-1,
                // then ensure tile kt's pair has landed (<=4 outstanding:
                // the pairs for kt+1 and kt+2 may still be in flight).
                const unsigned long long koff =
                    (unsigned long long)(kt + 2) * (BK * 2ull);
                int nb = buf + 2; if (nb >= NBUF) nb -= NBUF;
                tdm_load_tile(aBase + koff, (unsigned)nb * TILE_BYTES);
                tdm_load_tile(bBase + koff, (unsigned)(NBUF + nb) * TILE_BYTES);
                __builtin_amdgcn_s_wait_tensorcnt(4);
            } else if (kt + 1 < NK) {
                __builtin_amdgcn_s_wait_tensorcnt(2);
            } else {
                __builtin_amdgcn_s_wait_tensorcnt(0);
            }
        }
        __syncthreads();   // current buffer's tile is visible to all waves

        const __bf16* As = smem + lzero + buf * TILE_ELEMS;
        const __bf16* Bs = smem + lzero + (NBUF + buf) * TILE_ELEMS;

        // A fragment: lane(lrow,khalf) -> row, K = khalf*8 + {0..7, 16..23}
        BF16Frag afrag[4];
#pragma unroll
        for (int mt = 0; mt < 4; ++mt) {
            const int r = wm * 64 + mt * 16 + lrow;
            afrag[mt].h[0] = *(const v8bf*)(As + r * LDSP + khalf * 8);
            afrag[mt].h[1] = *(const v8bf*)(As + r * LDSP + khalf * 8 + 16);
        }
        // B fragment: lane(lrow,khalf) -> col, K = khalf*8 + {0..7, 16..23}
        BF16Frag bfrag[2];
#pragma unroll
        for (int nt = 0; nt < 2; ++nt) {
            const int c = wn * 32 + nt * 16 + lrow;
            bfrag[nt].h[0] = *(const v8bf*)(Bs + c * LDSP + khalf * 8);
            bfrag[nt].h[1] = *(const v8bf*)(Bs + c * LDSP + khalf * 8 + 16);
        }

#pragma unroll
        for (int mt = 0; mt < 4; ++mt)
#pragma unroll
            for (int nt = 0; nt < 2; ++nt)
                acc[mt][nt] = __builtin_amdgcn_wmma_f32_16x16x32_bf16(
                    false, afrag[mt].v, false, bfrag[nt].v,
                    (short)0, acc[mt][nt], false, false);

        __syncthreads();   // all waves done reading `buf`; safe to DMA over it
        buf = (buf + 1 == NBUF) ? 0 : buf + 1;
    }

    // Epilogue — C/D layout: elem j of lane l -> M = j + 8*(l>>4), N = l&15
#pragma unroll
    for (int mt = 0; mt < 4; ++mt) {
#pragma unroll
        for (int nt = 0; nt < 2; ++nt) {
#pragma unroll
            for (int j = 0; j < 8; ++j) {
                const int gr = rowBase + wm * 64 + mt * 16 + j + 8 * khalf;
                const int gc = colBase + wn * 32 + nt * 16 + lrow;
                const size_t idx = (size_t)gr * DDIM + gc;
                const float v = acc[mt][nt][j];
                if (mode == 0) {
                    Cf[idx] = v;
                } else if (mode == 1) {
                    const float p = Cf[idx];
                    Cf[idx] = v * p + p;
                } else if (mode == 2) {
                    const float p = Cf[idx];
                    const float r = v * p + p;
                    Cf[idx] = r;
                    obf[idx] = (__bf16)r;
                } else {
                    Xout[idx] = v + bias[gc];
                }
            }
        }
    }
}

extern "C" void kernel_launch(void* const* d_in, const int* in_sizes, int n_in,
                              void* d_out, int out_size, void* d_ws, size_t ws_size,
                              hipStream_t stream) {
    const float* z     = (const float*)d_in[0];
    const float* U     = (const float*)d_in[1];
    const float* masks = (const float*)d_in[2];
    const float* C_w   = (const float*)d_in[3];
    const float* C_b   = (const float*)d_in[4];
    float* xout = (float*)d_out;

    char* ws = (char*)d_ws;
    const size_t nZ = (size_t)NROW * DDIM;          // 16.7M
    const size_t nW = (size_t)6 * DDIM * DDIM;      // 25.2M
    const size_t nC = (size_t)DDIM * DDIM;          // 4.2M

    __bf16* zb   = (__bf16*)(ws);                                     // 32 MB
    __bf16* Wb   = (__bf16*)(ws + 2 * nZ);                            // 48 MB
    __bf16* Cwb  = (__bf16*)(ws + 2 * nZ + 2 * nW);                   // 8 MB
    float*  outf = (float*) (ws + 2 * nZ + 2 * nW + 2 * nC);          // 64 MB
    __bf16* obf  = (__bf16*)(ws + 2 * nZ + 2 * nW + 2 * nC + 4 * nZ); // 32 MB

    // Pre-passes: fold masks into U, convert operands to bf16.
    cvt_mask_mul_bf16<<<4096, 256, 0, stream>>>(U, masks, Wb, (long)nW);
    cvt_f32_to_bf16 <<<4096, 256, 0, stream>>>(z, zb, (long)nZ);
    cvt_f32_to_bf16 <<<2048, 256, 0, stream>>>(C_w, Cwb, (long)nC);

    dim3 grid(DDIM / BN, (int)(NROW / BM));  // (16, 64)
    dim3 block(256);

    // Degree 0: out = z @ W0^T
    gemm_chain_bf16<<<grid, block, 0, stream>>>(zb, Wb, outf, nullptr, nullptr, nullptr, 0);
    // Degrees 1..4: out = (z @ Wi^T) * out + out
    for (int i = 1; i < 5; ++i)
        gemm_chain_bf16<<<grid, block, 0, stream>>>(zb, Wb + (size_t)i * DDIM * DDIM,
                                                    outf, nullptr, nullptr, nullptr, 1);
    // Degree 5: same + emit bf16 copy for the final GEMM
    gemm_chain_bf16<<<grid, block, 0, stream>>>(zb, Wb + (size_t)5 * DDIM * DDIM,
                                                outf, obf, nullptr, nullptr, 2);
    // Final: x = out @ C_w^T + C_b
    gemm_chain_bf16<<<grid, block, 0, stream>>>(obf, Cwb, nullptr, nullptr, C_b, xout, 3);
}
